// AttentionMechanism_12721693130824
// MI455X (gfx1250) — matrix operand order repaired and checked by reading, not compiled, not run on hardware
//
#include <hip/hip_runtime.h>
#include <hip/hip_bf16.h>
#include <math.h>

// Problem constants (match reference)
constexpr int B   = 64;
constexpr int S   = 4096;
constexpr int DEC = 512;
constexpr int ENC = 256;

// Tiling
constexpr int CHUNK  = 128;            // S-rows per workgroup
constexpr int NCHUNK = S / CHUNK;      // 32 chunks per batch
constexpr int KSTEPS = ENC / 32;       // 8 wmma K-steps
constexpr int NTILES = ENC / 16;       // 16 N-tiles

typedef __attribute__((ext_vector_type(16))) __bf16 v16bf;
typedef __attribute__((ext_vector_type(8)))  float  v8f;

// Hardware tanh (CDNA5 TRANS op) when the builtin exists; ocml fallback.
__device__ __forceinline__ float fast_tanhf(float x) {
#if __has_builtin(__builtin_amdgcn_tanhf)
  return __builtin_amdgcn_tanhf(x);
#elif __has_builtin(__builtin_amdgcn_tanh_f32)
  return __builtin_amdgcn_tanh_f32(x);
#else
  return tanhf(x);
#endif
}

// ---------------------------------------------------------------------------
// Kernel 0a: pack We [ENC,ENC] fp32 into bf16 WMMA B-fragment order:
//   Bpack[t][ks][lane][e] = We[ks*32 + (lane>>4)*16 + e][t*16 + (lane&15)]
// ---------------------------------------------------------------------------
__global__ __launch_bounds__(256) void pack_We_kernel(
    const float* __restrict__ We, __bf16* __restrict__ Bpack) {
  int idx  = blockIdx.x * 256 + threadIdx.x;        // 0 .. 65535
  int e    = idx & 15;
  int lane = (idx >> 4) & 31;
  int ks   = (idx >> 9) & 7;
  int t    = idx >> 12;
  int n = t * 16 + (lane & 15);
  int k = ks * 32 + ((lane >> 4) * 16) + e;
  Bpack[idx] = (__bf16)We[k * ENC + n];
}

// ---------------------------------------------------------------------------
// Kernel 0b: dproj[b,n] = sum_k dh[b,k]*Wd[k,n] + bd[n] + be[n]   (16 MFLOP)
// ---------------------------------------------------------------------------
__global__ __launch_bounds__(256) void dproj_kernel(
    const float* __restrict__ dh, const float* __restrict__ Wd,
    const float* __restrict__ bd, const float* __restrict__ be,
    float* __restrict__ dproj) {
  int b = blockIdx.x;
  int n = threadIdx.x;
  float acc = bd[n] + be[n];
  #pragma unroll 8
  for (int k = 0; k < DEC; ++k)
    acc = fmaf(dh[b * DEC + k], Wd[k * ENC + n], acc);
  dproj[b * ENC + n] = acc;
}

// ---------------------------------------------------------------------------
// Main fused kernel: per (batch, chunk):
//   WMMA GEMM enc@We (+dproj as C seed) -> tanh -> dot Wa -> masked score
//   -> chunk softmax partials (m, l) and partial context pc[e]
// ---------------------------------------------------------------------------
__global__ __launch_bounds__(256) void attn_phaseA(
    const float*  __restrict__ enc,     // [B,S,ENC]
    const int*    __restrict__ mask,    // [B,S]
    const float*  __restrict__ Wa,      // [ENC]
    const float*  __restrict__ ba,      // [1]
    const float*  __restrict__ dproj,   // [B,ENC] (incl. bd+be)
    const __bf16* __restrict__ Bpack,   // [NTILES][KSTEPS][32][16]
    float* __restrict__ scores_ws,      // [B,S] masked raw scores
    float* __restrict__ pc_ws,          // [B,NCHUNK,ENC]
    float* __restrict__ mc_ws,          // [B,NCHUNK]
    float* __restrict__ lc_ws) {        // [B,NCHUNK]
  const int b     = blockIdx.y;
  const int chunk = blockIdx.x;
  const int s0    = chunk * CHUNK;
  const int tid   = threadIdx.x;
  const int wave  = tid >> 5;           // 0..7 -> M-tile
  const int lane  = tid & 31;
  const int lo    = lane & 15;
  const int hi    = lane >> 4;

  __shared__ float sh_scores[CHUNK];
  __shared__ float sh_p[CHUNK];
  __shared__ float sh_red[8];

  // ---- accumulators seeded with dproj (bias add for free via C operand) ----
  v8f acc[NTILES];
  #pragma unroll
  for (int t = 0; t < NTILES; ++t) {
    float c0 = dproj[b * ENC + t * 16 + lo];   // depends only on n
    v8f a;
    #pragma unroll
    for (int r = 0; r < 8; ++r) a[r] = c0;
    acc[t] = a;
  }

  // ---- main GEMM: rows s0+wave*16+lo, K=256 over 8 wmma steps ----
  const float* arow = enc + ((size_t)b * S + s0 + wave * 16 + lo) * ENC;
  for (int ks = 0; ks < KSTEPS; ++ks) {
    // A fragment: element e -> k = ks*32 + hi*16 + e
    const float* p0 = arow + ks * 32 + hi * 16;
    float4 f0 = *(const float4*)(p0 + 0);
    float4 f1 = *(const float4*)(p0 + 4);
    float4 f2 = *(const float4*)(p0 + 8);
    float4 f3 = *(const float4*)(p0 + 12);
    v16bf afrag;
    afrag[0]  = (__bf16)f0.x; afrag[1]  = (__bf16)f0.y;
    afrag[2]  = (__bf16)f0.z; afrag[3]  = (__bf16)f0.w;
    afrag[4]  = (__bf16)f1.x; afrag[5]  = (__bf16)f1.y;
    afrag[6]  = (__bf16)f1.z; afrag[7]  = (__bf16)f1.w;
    afrag[8]  = (__bf16)f2.x; afrag[9]  = (__bf16)f2.y;
    afrag[10] = (__bf16)f2.z; afrag[11] = (__bf16)f2.w;
    afrag[12] = (__bf16)f3.x; afrag[13] = (__bf16)f3.y;
    afrag[14] = (__bf16)f3.z; afrag[15] = (__bf16)f3.w;

    #pragma unroll
    for (int t = 0; t < NTILES; ++t) {
      v16bf bfrag = *(const v16bf*)(Bpack + (((t * KSTEPS + ks) * 32 + lane) << 4));
      acc[t] = __builtin_amdgcn_wmma_f32_16x16x32_bf16(
          /*neg_a=*/false, afrag, /*neg_b=*/false, bfrag,
          /*c_mod=*/(short)0, acc[t], /*reuse_a=*/false, /*reuse_b=*/false);
    }
  }

  // ---- energy = tanh(acc); score rows via Wa dot + lane reduction ----
  const float ba0 = ba[0];
  float sreg[8] = {0.f, 0.f, 0.f, 0.f, 0.f, 0.f, 0.f, 0.f};
  #pragma unroll
  for (int t = 0; t < NTILES; ++t) {
    float w = Wa[t * 16 + lo];
    #pragma unroll
    for (int r = 0; r < 8; ++r)
      sreg[r] = fmaf(fast_tanhf(acc[t][r]), w, sreg[r]);
  }
  #pragma unroll
  for (int r = 0; r < 8; ++r) {
    float v = sreg[r];
    v += __shfl_xor(v, 1, 32);
    v += __shfl_xor(v, 2, 32);
    v += __shfl_xor(v, 4, 32);
    v += __shfl_xor(v, 8, 32);            // hi-groups stay separate (bit 4)
    if (lo == 0) {
      int sl = wave * 16 + r + 8 * hi;    // row within chunk
      float sc = v + ba0;
      if (mask[(size_t)b * S + s0 + sl] == 0) sc = -1e10f;
      sh_scores[sl] = sc;
      scores_ws[(size_t)b * S + s0 + sl] = sc;
    }
  }
  __syncthreads();

  // ---- chunk max ----
  float lm = (tid < CHUNK) ? sh_scores[tid] : -3.0e38f;
  #pragma unroll
  for (int o = 16; o >= 1; o >>= 1) lm = fmaxf(lm, __shfl_xor(lm, o, 32));
  if (lane == 0) sh_red[wave] = lm;
  __syncthreads();
  float M = sh_red[0];
  #pragma unroll
  for (int w2 = 1; w2 < 8; ++w2) M = fmaxf(M, sh_red[w2]);

  // ---- p = exp(score - M), chunk sum ----
  float pe = 0.f;
  if (tid < CHUNK) { pe = __expf(sh_scores[tid] - M); sh_p[tid] = pe; }
  __syncthreads();                         // sh_red consumed; sh_p published
  float ls = pe;
  #pragma unroll
  for (int o = 16; o >= 1; o >>= 1) ls += __shfl_xor(ls, o, 32);
  if (lane == 0) sh_red[wave] = ls;
  __syncthreads();
  float Lc = sh_red[0] + sh_red[1] + sh_red[2] + sh_red[3] +
             sh_red[4] + sh_red[5] + sh_red[6] + sh_red[7];

  // ---- partial context pc[e] = sum_s p_s * enc[b,s,e]  (chunk hot in L2) ----
  const float* encb = enc + ((size_t)b * S + s0) * ENC + tid;
  float pcv = 0.f;
  #pragma unroll 4
  for (int s2 = 0; s2 < CHUNK; ++s2)
    pcv = fmaf(sh_p[s2], encb[(size_t)s2 * ENC], pcv);
  pc_ws[((size_t)b * NCHUNK + chunk) * ENC + tid] = pcv;
  if (tid == 0) {
    mc_ws[b * NCHUNK + chunk] = M;
    lc_ws[b * NCHUNK + chunk] = Lc;
  }
}

// ---------------------------------------------------------------------------
// Combine per-chunk partials (flash-attention style) -> context, global M/L
// ---------------------------------------------------------------------------
__global__ __launch_bounds__(256) void attn_combine(
    const float* __restrict__ pc_ws, const float* __restrict__ mc_ws,
    const float* __restrict__ lc_ws, float* __restrict__ ctx_out,
    float* __restrict__ ML_ws) {
  int b = blockIdx.x;
  int tid = threadIdx.x;                   // 0..255 = ENC column
  __shared__ float shm[NCHUNK];
  __shared__ float shL;
  if (tid < NCHUNK) shm[tid] = mc_ws[b * NCHUNK + tid];
  __syncthreads();
  float M = -3.0e38f;
  #pragma unroll
  for (int c = 0; c < NCHUNK; ++c) M = fmaxf(M, shm[c]);
  if (tid == 0) {
    float L = 0.f;
    for (int c = 0; c < NCHUNK; ++c)
      L = fmaf(lc_ws[b * NCHUNK + c], __expf(shm[c] - M), L);
    shL = L;
  }
  __syncthreads();
  float L = shL;
  float acc = 0.f;
  #pragma unroll
  for (int c = 0; c < NCHUNK; ++c)
    acc = fmaf(__expf(shm[c] - M), pc_ws[((size_t)b * NCHUNK + c) * ENC + tid], acc);
  ctx_out[b * ENC + tid] = acc / L;
  if (tid == 0) { ML_ws[b] = M; ML_ws[B + b] = L; }
}

// ---------------------------------------------------------------------------
// attn[b,s] = exp(score - M_b) / L_b
// ---------------------------------------------------------------------------
__global__ __launch_bounds__(256) void attn_normalize(
    const float* __restrict__ scores_ws, const float* __restrict__ ML_ws,
    float* __restrict__ attn_out) {
  int idx = blockIdx.x * 256 + threadIdx.x;   // 0 .. B*S-1
  int b = idx >> 12;                          // S = 4096
  float M = ML_ws[b];
  float L = ML_ws[B + b];
  attn_out[idx] = __expf(scores_ws[idx] - M) / L;
}

// ---------------------------------------------------------------------------
extern "C" void kernel_launch(void* const* d_in, const int* in_sizes, int n_in,
                              void* d_out, int out_size, void* d_ws, size_t ws_size,
                              hipStream_t stream) {
  const float* dh   = (const float*)d_in[0];  // [B,1,DEC]
  const float* enc  = (const float*)d_in[1];  // [B,S,ENC]
  const int*   mask = (const int*)  d_in[2];  // [B,S]
  const float* Wd   = (const float*)d_in[3];  // [DEC,ENC]
  const float* bd   = (const float*)d_in[4];  // [ENC]
  const float* We   = (const float*)d_in[5];  // [ENC,ENC]
  const float* be   = (const float*)d_in[6];  // [ENC]
  const float* Wa   = (const float*)d_in[7];  // [ENC]
  const float* ba   = (const float*)d_in[8];  // [1]

  float* ctx_out  = (float*)d_out;            // [B*ENC]
  float* attn_out = (float*)d_out + B * ENC;  // [B*S]

  // Workspace layout (~3.4 MB total)
  char* ws = (char*)d_ws;
  size_t off = 0;
  __bf16* Bpack  = (__bf16*)(ws + off); off += (size_t)ENC * ENC * 2;        // 128 KB
  float*  dproj  = (float*) (ws + off); off += (size_t)B * ENC * 4;          // 64 KB
  float*  scores = (float*) (ws + off); off += (size_t)B * S * 4;            // 1 MB
  float*  pc     = (float*) (ws + off); off += (size_t)B * NCHUNK * ENC * 4; // 2 MB
  float*  mc     = (float*) (ws + off); off += (size_t)B * NCHUNK * 4;
  float*  lc     = (float*) (ws + off); off += (size_t)B * NCHUNK * 4;
  float*  ML     = (float*) (ws + off); off += (size_t)2 * B * 4;

  pack_We_kernel<<<ENC * ENC / 256, 256, 0, stream>>>(We, Bpack);
  dproj_kernel<<<B, ENC, 0, stream>>>(dh, Wd, bd, be, dproj);
  attn_phaseA<<<dim3(NCHUNK, B), 256, 0, stream>>>(
      enc, mask, Wa, ba, dproj, Bpack, scores, pc, mc, lc);
  attn_combine<<<B, ENC, 0, stream>>>(pc, mc, lc, ctx_out, ML);
  attn_normalize<<<B * S / 256, 256, 0, stream>>>(scores, ML, attn_out);
}
